// InfoGeoNet_73323681677675
// MI455X (gfx1250) — matrix operand order repaired and checked by reading, not compiled, not run on hardware
//
#include <hip/hip_runtime.h>
#include <hip/hip_bf16.h>

// ---------------------------------------------------------------------------
// Types for CDNA5 WMMA
// ---------------------------------------------------------------------------
typedef __attribute__((ext_vector_type(16))) __bf16 v16bf;
typedef __attribute__((ext_vector_type(8)))  float  v8f;

static __device__ __forceinline__ unsigned short f2bf(float f) {
    // round-to-nearest-even fp32 -> bf16 (finite inputs)
    unsigned int u = __float_as_uint(f);
    unsigned int lsb = (u >> 16) & 1u;
    u += 0x7fffu + lsb;
    return (unsigned short)(u >> 16);
}

// ---------------------------------------------------------------------------
// fp32 -> bf16 weight conversion
// ---------------------------------------------------------------------------
__global__ void cvt_bf16_kernel(const float* __restrict__ in,
                                unsigned short* __restrict__ out, int n) {
    int t = blockIdx.x * 256 + threadIdx.x;
    if (t < n) out[t] = f2bf(in[t]);
}

// ---------------------------------------------------------------------------
// Implicit-GEMM 3x3 conv (pad=1) using v_wmma_f32_16x16x32_bf16.
// Block: 128 threads = 4 waves. Tile: M=64 output pixels x N=32 out-channels.
// Each wave owns 16 M-rows and both 16-wide N tiles (A fragment reused,
// 2 WMMA per K-step). K = (C1+C2)*9, stepped by 32, zero padded.
// Fragments are loaded from LDS as contiguous 128-bit chunks.
// Epilogue goes through an LDS transpose so global stores are 16 contiguous
// pixels (64 B) per thread.
// act: 0 = none, 1 = relu, 2 = BN(eval, var=1,mean=0) + exact GELU
// ---------------------------------------------------------------------------
#define TILE_M 64
#define TILE_K 32
#define TILE_N 32

__global__ __launch_bounds__(128)
void conv3x3_wmma_kernel(const float* __restrict__ in,
                         const float* __restrict__ in2,
                         const unsigned short* __restrict__ wbf, // [Cout][K] bf16
                         const float* __restrict__ bias,
                         const float* __restrict__ bn_g,
                         const float* __restrict__ bn_b,
                         float* __restrict__ out,
                         int C1, int C2, int Hin, int Win,
                         int Hout, int Wout, int Cout,
                         int stride, int act)
{
    const int b      = blockIdx.z;
    const int coBase = blockIdx.y * TILE_N;
    const int tileM  = blockIdx.x * TILE_M;
    const int tid    = threadIdx.x;
    const int wave   = tid >> 5;
    const int lane   = tid & 31;

    const int M   = Hout * Wout;
    const int Cin = C1 + C2;
    const int K   = Cin * 9;

    __shared__ unsigned short As[TILE_M][TILE_K];       // [m][k]
    __shared__ unsigned short Bst[TILE_N][TILE_K];      // transposed: [n][k]
    __shared__ float          Ds[TILE_M][TILE_N + 2];   // epilogue transpose

    union { v8f v; float f[8]; } acc0, acc1;
#pragma unroll
    for (int i = 0; i < 8; ++i) { acc0.f[i] = 0.0f; acc1.f[i] = 0.0f; }

    // ---- per-thread staging geometry (hoisted) ----
    // A: each thread fills 16 contiguous k slots of one row.
    const int aRow   = tid >> 1;            // 0..63
    const int aKoff  = (tid & 1) * 16;      // 0 or 16
    const int pA     = tileM + aRow;
    const int oy     = pA / Wout;
    const int ox     = pA - oy * Wout;
    const bool pOk   = (pA < M);
    // B: each thread fills 8 contiguous k slots of one n row.
    const int bN     = tid >> 2;            // 0..31
    const int bKoff  = (tid & 3) * 8;       // 0,8,16,24
    const unsigned short* wRow = wbf + (long)(coBase + bN) * K;

    for (int k0 = 0; k0 < K; k0 += TILE_K) {
        // ---- stage A tile (64x32 bf16): 16 contiguous k per thread ----
        {
            int k   = k0 + aKoff;
            int ci  = k / 9;
            int rem = k - ci * 9;
            int ky  = rem / 3;
            int kx  = rem - ky * 3;
            unsigned short* dst = &As[aRow][aKoff];
#pragma unroll
            for (int e = 0; e < 16; ++e) {
                float val = 0.0f;
                if (pOk && (k + e) < K) {
                    int iy = oy * stride + ky - 1;
                    int ix = ox * stride + kx - 1;
                    if (iy >= 0 && iy < Hin && ix >= 0 && ix < Win) {
                        val = (ci < C1)
                            ? in [(((long)b * C1 + ci)        * Hin + iy) * Win + ix]
                            : in2[(((long)b * C2 + (ci - C1)) * Hin + iy) * Win + ix];
                    }
                }
                dst[e] = f2bf(val);
                // incremental (kx,ky,ci) walk
                if (++kx == 3) { kx = 0; if (++ky == 3) { ky = 0; ++ci; } }
            }
        }
        // ---- stage B tile transposed (32n x 32k bf16): 8 contiguous k ----
        {
            unsigned short* dst = &Bst[bN][bKoff];
            int k = k0 + bKoff;
#pragma unroll
            for (int e = 0; e < 8; ++e)
                dst[e] = ((k + e) < K) ? wRow[k + e] : (unsigned short)0;
            // prefetch next K-step weight chunk into L2/L0
            if (k0 + TILE_K < K)
                __builtin_prefetch(&wRow[k0 + TILE_K + bKoff], 0, 1);
        }
        __syncthreads();

        // ---- fragment loads (contiguous 128-bit LDS reads) ----
        union { v16bf v; unsigned short u[16]; uint4 q[2]; } af, bf0, bf1;
        const int mLoc = wave * 16 + (lane & 15);
        const int kA   = (lane < 16) ? 0 : 8;
        af.q[0] = *(const uint4*)&As[mLoc][kA];        // K = kA .. kA+7
        af.q[1] = *(const uint4*)&As[mLoc][kA + 16];   // K = kA+16 .. kA+23
        const int nB = lane & 15;
        const int kB = (lane < 16) ? 0 : 8;
        bf0.q[0] = *(const uint4*)&Bst[nB][kB];
        bf0.q[1] = *(const uint4*)&Bst[nB][kB + 16];
        bf1.q[0] = *(const uint4*)&Bst[16 + nB][kB];
        bf1.q[1] = *(const uint4*)&Bst[16 + nB][kB + 16];

        acc0.v = __builtin_amdgcn_wmma_f32_16x16x32_bf16(
            false, af.v, false, bf0.v, (short)0, acc0.v, false, false);
        acc1.v = __builtin_amdgcn_wmma_f32_16x16x32_bf16(
            false, af.v, false, bf1.v, (short)0, acc1.v, false, false);
        __syncthreads();
    }

    // ---- epilogue: D -> LDS transpose -> coalesced stores ----
#pragma unroll
    for (int r = 0; r < 8; ++r) {
        int m = wave * 16 + r + ((lane < 16) ? 0 : 8);
        Ds[m][lane & 15]        = acc0.f[r];
        Ds[m][16 + (lane & 15)] = acc1.f[r];
    }
    __syncthreads();

    {
        const int n    = tid >> 2;           // 0..31
        const int co   = coBase + n;
        const int pseg = (tid & 3) * 16;     // 0,16,32,48
        const float bv = bias[co];
        float bsc = 1.0f, bsh = 0.0f;
        if (act == 2) {
            bsc = bn_g[co] * 0.99999500003749969f; // g / sqrt(1 + 1e-5)
            bsh = bn_b[co];
        }
        float* orow = out + ((long)b * Cout + co) * M + tileM + pseg;
#pragma unroll
        for (int j = 0; j < 16; ++j) {
            int p = tileM + pseg + j;
            if (p < M) {
                float v = Ds[pseg + j][n] + bv;
                if (act == 1) {
                    v = fmaxf(v, 0.0f);
                } else if (act == 2) {
                    v = v * bsc + bsh;
                    v = 0.5f * v * (1.0f + erff(v * 0.70710678118654752f));
                }
                orow[j] = v;
            }
        }
    }
}

// ---------------------------------------------------------------------------
// 1x1 conv (tiny N). mode: 0 = linear, 1 = sigmoid
// ---------------------------------------------------------------------------
__global__ void conv1x1_kernel(const float* __restrict__ in,
                               const float* __restrict__ w,
                               const float* __restrict__ bias,
                               float* __restrict__ out,
                               int Cin, int Cout, int Npix, int B, int mode)
{
    long t = (long)blockIdx.x * blockDim.x + threadIdx.x;
    long total = (long)B * Cout * Npix;
    if (t >= total) return;
    int p = (int)(t % Npix);
    long r = t / Npix;
    int o = (int)(r % Cout);
    int b = (int)(r / Cout);
    const float* ip = in + (long)b * Cin * Npix + p;
    float s = bias[o];
    for (int c = 0; c < Cin; ++c) s += ip[(long)c * Npix] * w[o * Cin + c];
    if (mode == 1) s = 1.0f / (1.0f + expf(-s));
    out[t] = s;
}

// ---------------------------------------------------------------------------
// Fisher map: sum_c max(p(1-p), eps), p = sigmoid(logit/clip(T,0.1,10))
// ---------------------------------------------------------------------------
__global__ void fisher_kernel(const float* __restrict__ logits,
                              const float* __restrict__ temp_ptr,
                              float* __restrict__ fout, int Npix, int B)
{
    long t = (long)blockIdx.x * blockDim.x + threadIdx.x;
    long total = (long)B * Npix;
    if (t >= total) return;
    int p = (int)(t % Npix);
    int b = (int)(t / Npix);
    float temp = fminf(fmaxf(temp_ptr[0], 0.1f), 10.0f);
    float s = 0.0f;
    for (int c = 0; c < 15; ++c) {
        float z  = logits[((long)b * 15 + c) * Npix + p] / temp;
        float pr = 1.0f / (1.0f + expf(-z));
        s += fmaxf(pr * (1.0f - pr), 1e-6f);
    }
    fout[t] = s;
}

// ---------------------------------------------------------------------------
// Top-100 per batch (iterative argmax, ties -> lower index). Destroys `sc`.
// ---------------------------------------------------------------------------
__global__ __launch_bounds__(256)
void topk_kernel(float* __restrict__ sc, float* __restrict__ top_scores,
                 int* __restrict__ top_idx, int Npix)
{
    int b = blockIdx.x;
    float* s = sc + (long)b * Npix;
    __shared__ float sval[256];
    __shared__ int   sidx[256];
    for (int it = 0; it < 100; ++it) {
        float bv = -3.402823466e38f;
        int   bi = 0x7fffffff;
        for (int i = threadIdx.x; i < Npix; i += 256) {
            float v = s[i];
            if (v > bv || (v == bv && i < bi)) { bv = v; bi = i; }
        }
        sval[threadIdx.x] = bv;
        sidx[threadIdx.x] = bi;
        __syncthreads();
        for (int off = 128; off > 0; off >>= 1) {
            if (threadIdx.x < off) {
                float ov = sval[threadIdx.x + off];
                int   oi = sidx[threadIdx.x + off];
                if (ov > sval[threadIdx.x] ||
                    (ov == sval[threadIdx.x] && oi < sidx[threadIdx.x])) {
                    sval[threadIdx.x] = ov;
                    sidx[threadIdx.x] = oi;
                }
            }
            __syncthreads();
        }
        if (threadIdx.x == 0) {
            top_scores[b * 100 + it] = sval[0];
            top_idx[b * 100 + it]    = sidx[0];
            s[sidx[0]] = -3.402823466e38f;
        }
        __syncthreads();
    }
}

// ---------------------------------------------------------------------------
// Per-proposal detection heads. One 128-thread block per proposal.
// ---------------------------------------------------------------------------
__global__ __launch_bounds__(128)
void heads_kernel(const float* __restrict__ features, // [B][128][Npix]
                  const int*   __restrict__ top_idx,  // [B*100]
                  const float* __restrict__ top_scores,
                  const float* __restrict__ det_w,  const float* __restrict__ det_b,
                  const float* __restrict__ co_w1,  const float* __restrict__ co_b1,
                  const float* __restrict__ co_w2,  const float* __restrict__ co_b2,
                  const float* __restrict__ sz_w1,  const float* __restrict__ sz_b1,
                  const float* __restrict__ sz_w2,  const float* __restrict__ sz_b2,
                  const float* __restrict__ an_w1,  const float* __restrict__ an_b1,
                  const float* __restrict__ an_w2,  const float* __restrict__ an_b2,
                  const float* __restrict__ cls_w1, const float* __restrict__ cls_b1,
                  const float* __restrict__ cls_w2, const float* __restrict__ cls_b2,
                  const float* __restrict__ conf_w1,const float* __restrict__ conf_b1,
                  const float* __restrict__ conf_w2,const float* __restrict__ conf_b2,
                  float* __restrict__ centers, float* __restrict__ wh,
                  float* __restrict__ angles,  float* __restrict__ cls,
                  float* __restrict__ conf,
                  int Npix)
{
    const int pid = blockIdx.x;        // 0..799
    const int b   = pid / 100;
    const int tid = threadIdx.x;

    __shared__ float f[128];
    __shared__ float ft[128];
    __shared__ float h[64];

    const int   idx   = top_idx[pid];
    const float score = top_scores[pid];

    // gather local feature vector
    f[tid] = features[((long)b * 128 + tid) * Npix + idx];
    __syncthreads();

    // det: feat = relu(det_w @ f + det_b)
    {
        float s = det_b[tid];
        for (int i = 0; i < 128; ++i) s += f[i] * det_w[tid * 128 + i];
        ft[tid] = fmaxf(s, 0.0f);
    }
    __syncthreads();

    // ---- centers head (co) ----
    if (tid < 32) {
        float s = co_b1[tid];
        for (int i = 0; i < 128; ++i) s += ft[i] * co_w1[tid * 128 + i];
        h[tid] = fmaxf(s, 0.0f);
    }
    __syncthreads();
    if (tid < 2) {
        float s = co_b2[tid];
        for (int i = 0; i < 32; ++i) s += h[i] * co_w2[tid * 32 + i];
        float off = tanhf(s);
        int y = idx / 160;
        int x = idx - y * 160;
        float coord = (tid == 0) ? ((float)x / 160.0f) : ((float)y / 160.0f);
        centers[pid * 2 + tid] = (coord + off * (1.0f / 160.0f)) * 640.0f;
    }
    __syncthreads();

    // ---- size head (sz) ----
    if (tid < 32) {
        float s = sz_b1[tid];
        for (int i = 0; i < 128; ++i) s += ft[i] * sz_w1[tid * 128 + i];
        h[tid] = fmaxf(s, 0.0f);
    }
    __syncthreads();
    if (tid < 2) {
        float s = sz_b2[tid];
        for (int i = 0; i < 32; ++i) s += h[i] * sz_w2[tid * 32 + i];
        s = fminf(fmaxf(s, -5.0f), 5.0f);
        wh[pid * 2 + tid] = expf(s) * 64.0f;   // IMG / 10
    }
    __syncthreads();

    // ---- angle head (an) ----
    if (tid < 16) {
        float s = an_b1[tid];
        for (int i = 0; i < 128; ++i) s += ft[i] * an_w1[tid * 128 + i];
        h[tid] = fmaxf(s, 0.0f);
    }
    __syncthreads();
    if (tid == 0) {
        float a = an_b2[0];
        for (int i = 0; i < 16; ++i) a += h[i] * an_w2[i];
        angles[pid] = atan2f(sinf(a), cosf(a));
    }
    __syncthreads();

    // ---- class head (cls) ----
    if (tid < 64) {
        float s = cls_b1[tid];
        for (int i = 0; i < 128; ++i) s += ft[i] * cls_w1[tid * 128 + i];
        h[tid] = fmaxf(s, 0.0f);
    }
    __syncthreads();
    if (tid < 15) {
        float s = cls_b2[tid];
        for (int i = 0; i < 64; ++i) s += h[i] * cls_w2[tid * 64 + i];
        cls[pid * 15 + tid] = s;
    }
    __syncthreads();

    // ---- confidence head (conf), input = [feat(128), top_score] ----
    if (tid < 16) {
        float s = conf_b1[tid];
        for (int i = 0; i < 128; ++i) s += ft[i] * conf_w1[tid * 129 + i];
        s += score * conf_w1[tid * 129 + 128];
        h[tid] = fmaxf(s, 0.0f);
    }
    __syncthreads();
    if (tid == 0) {
        float s = conf_b2[0];
        for (int i = 0; i < 16; ++i) s += h[i] * conf_w2[i];
        conf[pid] = 1.0f / (1.0f + expf(-s));
    }
}

// ---------------------------------------------------------------------------
// Host-side orchestration
// ---------------------------------------------------------------------------
extern "C" void kernel_launch(void* const* d_in, const int* in_sizes, int n_in,
                              void* d_out, int out_size, void* d_ws, size_t ws_size,
                              hipStream_t stream)
{
    (void)in_sizes; (void)n_in; (void)out_size; (void)ws_size;

    // ---- inputs ----
    const float* images = (const float*)d_in[0];
    const float* enc_w1 = (const float*)d_in[1];
    const float* enc_b1 = (const float*)d_in[2];
    const float* bn1_g  = (const float*)d_in[3];
    const float* bn1_b  = (const float*)d_in[4];
    const float* enc_w2 = (const float*)d_in[5];
    const float* enc_b2 = (const float*)d_in[6];
    const float* bn2_g  = (const float*)d_in[7];
    const float* bn2_b  = (const float*)d_in[8];
    const float* enc_w3 = (const float*)d_in[9];
    const float* enc_b3 = (const float*)d_in[10];
    const float* bn3_g  = (const float*)d_in[11];
    const float* bn3_b  = (const float*)d_in[12];
    const float* dist_w1 = (const float*)d_in[13];
    const float* dist_b1 = (const float*)d_in[14];
    const float* dist_w2 = (const float*)d_in[15];
    const float* dist_b2 = (const float*)d_in[16];
    const float* feat_w  = (const float*)d_in[17];
    const float* feat_b  = (const float*)d_in[18];
    const float* bnf_g   = (const float*)d_in[19];
    const float* bnf_b   = (const float*)d_in[20];
    const float* temperature = (const float*)d_in[21];
    const float* score_w1 = (const float*)d_in[22];
    const float* score_b1 = (const float*)d_in[23];
    const float* score_w2 = (const float*)d_in[24];
    const float* score_b2 = (const float*)d_in[25];

    // ---- output layout (floats) ----
    float* o       = (float*)d_out;
    float* centers = o + 0;       // 8*100*2  = 1600
    float* whp     = o + 1600;    // 1600
    float* ang     = o + 3200;    // 800
    float* clsp    = o + 4000;    // 12000
    float* confp   = o + 16000;   // 800
    float* tscore  = o + 16800;   // 800
    float* fisher  = o + 17600;   // 8*160*160 = 204800

    // ---- workspace layout (bytes) ----
    char* ws = (char*)d_ws;
    float* bufA = (float*)(ws + 0);          // x1 (8x32x320x320) -> later features (8x128x160x160)
    float* bufB = (float*)(ws + 104857600);  // x2 (8x64x160x160)
    float* bufC = (float*)(ws + 157286400);  // enc (8x128x160x160)
    float* bufE = (float*)(ws + 262144000);  // dist mid / score mid (8x64x160x160)
    float* bufF = (float*)(ws + 314572800);  // logits (8x15x160x160)
    float* bufG = (float*)(ws + 326860800);  // scores_map (8x160x160), destroyed by topk
    unsigned short* wbf = (unsigned short*)(ws + 327680000); // bf16 weights (388512 elems)
    int* topIdx = (int*)(ws + 328458240);    // 800 ints
    // total workspace used: ~328.5 MB

    // bf16 weight offsets (elements)
    const long W1 = 0;       // 32*27   = 864
    const long W2 = 864;     // 64*288  = 18432
    const long W3 = 19296;   // 128*576 = 73728
    const long WD1 = 93024;  // 64*1152 = 73728
    const long WF  = 166752; // 128*1152= 147456
    const long WS1 = 314208; // 64*1161 = 74304

    // ---- convert conv weights to bf16 ----
    cvt_bf16_kernel<<<(864 + 255) / 256, 256, 0, stream>>>(enc_w1, wbf + W1, 864);
    cvt_bf16_kernel<<<(18432 + 255) / 256, 256, 0, stream>>>(enc_w2, wbf + W2, 18432);
    cvt_bf16_kernel<<<(73728 + 255) / 256, 256, 0, stream>>>(enc_w3, wbf + W3, 73728);
    cvt_bf16_kernel<<<(73728 + 255) / 256, 256, 0, stream>>>(dist_w1, wbf + WD1, 73728);
    cvt_bf16_kernel<<<(147456 + 255) / 256, 256, 0, stream>>>(feat_w, wbf + WF, 147456);
    cvt_bf16_kernel<<<(74304 + 255) / 256, 256, 0, stream>>>(score_w1, wbf + WS1, 74304);

    const int B = 8;
    const int Npix = 160 * 160;

    // ---- encoder conv1: 3 -> 32, stride 2, 640 -> 320, BN+GELU ----
    {
        dim3 g(102400 / TILE_M, 32 / TILE_N, B);
        conv3x3_wmma_kernel<<<g, 128, 0, stream>>>(
            images, nullptr, wbf + W1, enc_b1, bn1_g, bn1_b, bufA,
            3, 0, 640, 640, 320, 320, 32, 2, 2);
    }
    // ---- encoder conv2: 32 -> 64, stride 2, 320 -> 160, BN+GELU ----
    {
        dim3 g(Npix / TILE_M, 64 / TILE_N, B);
        conv3x3_wmma_kernel<<<g, 128, 0, stream>>>(
            bufA, nullptr, wbf + W2, enc_b2, bn2_g, bn2_b, bufB,
            32, 0, 320, 320, 160, 160, 64, 2, 2);
    }
    // ---- encoder conv3: 64 -> 128, stride 1, BN+GELU -> enc ----
    {
        dim3 g(Npix / TILE_M, 128 / TILE_N, B);
        conv3x3_wmma_kernel<<<g, 128, 0, stream>>>(
            bufB, nullptr, wbf + W3, enc_b3, bn3_g, bn3_b, bufC,
            64, 0, 160, 160, 160, 160, 128, 1, 2);
    }
    // ---- dist conv1: 128 -> 64, ReLU ----
    {
        dim3 g(Npix / TILE_M, 64 / TILE_N, B);
        conv3x3_wmma_kernel<<<g, 128, 0, stream>>>(
            bufC, nullptr, wbf + WD1, dist_b1, nullptr, nullptr, bufE,
            128, 0, 160, 160, 160, 160, 64, 1, 1);
    }
    // ---- dist conv2 (1x1): 64 -> 15 -> logits ----
    {
        long total = (long)B * 15 * Npix;
        conv1x1_kernel<<<(unsigned)((total + 255) / 256), 256, 0, stream>>>(
            bufE, dist_w2, dist_b2, bufF, 64, 15, Npix, B, 0);
    }
    // ---- feat conv: 128 -> 128, BN+GELU -> features (reuses bufA) ----
    {
        dim3 g(Npix / TILE_M, 128 / TILE_N, B);
        conv3x3_wmma_kernel<<<g, 128, 0, stream>>>(
            bufC, nullptr, wbf + WF, feat_b, bnf_g, bnf_b, bufA,
            128, 0, 160, 160, 160, 160, 128, 1, 2);
    }
    // ---- fisher map -> d_out ----
    {
        long total = (long)B * Npix;
        fisher_kernel<<<(unsigned)((total + 255) / 256), 256, 0, stream>>>(
            bufF, temperature, fisher, Npix, B);
    }
    // ---- score conv1: [features(128) ++ fisher(1)] -> 64, ReLU ----
    {
        dim3 g(Npix / TILE_M, 64 / TILE_N, B);
        conv3x3_wmma_kernel<<<g, 128, 0, stream>>>(
            bufA, fisher, wbf + WS1, score_b1, nullptr, nullptr, bufE,
            128, 1, 160, 160, 160, 160, 64, 1, 1);
    }
    // ---- score conv2 (1x1) + sigmoid -> scores_map ----
    {
        long total = (long)B * Npix;
        conv1x1_kernel<<<(unsigned)((total + 255) / 256), 256, 0, stream>>>(
            bufE, score_w2, score_b2, bufG, 64, 1, Npix, B, 1);
    }
    // ---- top-100 per batch (destroys bufG) ----
    topk_kernel<<<B, 256, 0, stream>>>(bufG, tscore, topIdx, Npix);

    // ---- detection heads ----
    heads_kernel<<<B * 100, 128, 0, stream>>>(
        bufA, topIdx, tscore,
        (const float*)d_in[26], (const float*)d_in[27],
        (const float*)d_in[28], (const float*)d_in[29],
        (const float*)d_in[30], (const float*)d_in[31],
        (const float*)d_in[32], (const float*)d_in[33],
        (const float*)d_in[34], (const float*)d_in[35],
        (const float*)d_in[36], (const float*)d_in[37],
        (const float*)d_in[38], (const float*)d_in[39],
        (const float*)d_in[40], (const float*)d_in[41],
        (const float*)d_in[42], (const float*)d_in[43],
        (const float*)d_in[44], (const float*)d_in[45],
        (const float*)d_in[46], (const float*)d_in[47],
        centers, whp, ang, clsp, confp,
        Npix);
}
